// MultiHeadAttn_65025804862093
// MI455X (gfx1250) — matrix operand order repaired and checked by reading, not compiled, not run on hardware
//
#include <hip/hip_runtime.h>
#include <hip/hip_bf16.h>

// ---------------- types ----------------
typedef __attribute__((ext_vector_type(16))) __bf16         v16bf;
typedef __attribute__((ext_vector_type(8)))  float          v8f;
typedef __attribute__((ext_vector_type(8)))  unsigned short ush8;
typedef __attribute__((ext_vector_type(4)))  unsigned short ush4;
typedef __attribute__((ext_vector_type(4)))  float          f4;
typedef __attribute__((ext_vector_type(4)))  unsigned int   u32x4;
typedef __attribute__((ext_vector_type(8)))  int            i32x8;
typedef __attribute__((ext_vector_type(4)))  int            i32x4;

union FragBF16 {            // 16 bf16 = one WMMA A/B operand (8 VGPRs)
    ush8  u[2];
    v16bf b;
};

// Problem constants (fixed by the reference)
#define BATCH 4
#define SEQ   2048
#define DMODEL 1024
#define NHEAD 16
#define DHEAD 64
#define MROWS (BATCH * SEQ)        // 8192

__device__ __forceinline__ unsigned short f2bf(float x) {
    unsigned int u = __float_as_uint(x);
    unsigned int r = u + 0x7FFFu + ((u >> 16) & 1u);   // round-to-nearest-even
    return (unsigned short)(r >> 16);
}

// LDS offset of a __shared__ pointer: flat LDS aperture address maps
// to LDS (ISA 10.2), so the low 32 bits are the LDS offset.
__device__ __forceinline__ unsigned lds_off32(const void* p) {
    return (unsigned)(unsigned long long)p;
}

// ---- Tensor Data Mover: async DMA of a 2-D bf16 tile (rows x cols) from
// global (row pitch `stride` elems) into LDS (packed, pitch = cols).
// D# layout per CDNA5 ISA; clang-23 6-arg builtin:
// (u32x4 g0, i32x8 g1, i32x4 g2, i32x4 g3, i32x8 g4, i32 cpol).
__device__ __forceinline__ void tdm_load_2d(unsigned lds_addr,
                                            unsigned long long gaddr,
                                            unsigned tensor_d0, unsigned tensor_d1,
                                            unsigned tile_d0, unsigned tile_d1,
                                            unsigned long long stride_e) {
    u32x4 g0;
    g0[0] = 1u;                                            // count=1, user mode
    g0[1] = lds_addr;                                      // lds_addr
    g0[2] = (unsigned)gaddr;                               // global_addr[31:0]
    g0[3] = (unsigned)((gaddr >> 32) & 0x01FFFFFFu)        // global_addr[56:32]
          | (2u << 30);                                    // type=2 ("image")
    i32x8 g1;
    g1[0] = (int)(1u << 16);                               // data_size=1 (2 bytes)
    g1[1] = (int)((tensor_d0 & 0xFFFFu) << 16);            // tensor_dim0[15:0]
    g1[2] = (int)((tensor_d0 >> 16) | ((tensor_d1 & 0xFFFFu) << 16));
    g1[3] = (int)((tensor_d1 >> 16) | (tile_d0 << 16));    // tile_dim0
    g1[4] = (int)tile_d1;                                  // tile_dim1 (tile_dim2=0)
    g1[5] = (int)(unsigned)stride_e;                       // tensor_dim0_stride[31:0]
    g1[6] = (int)((stride_e >> 32) & 0xFFFFu);             // stride[47:32]
    g1[7] = 0;
    i32x4 z4 = {0, 0, 0, 0};
    i32x8 z8 = {0, 0, 0, 0, 0, 0, 0, 0};
    __builtin_amdgcn_tensor_load_to_lds(g0, g1, z4, z4, z8, 0);
}

// ---------------- fp32 -> bf16 convert ----------------
__global__ void mha_cvt_bf16(const float* __restrict__ src,
                             unsigned short* __restrict__ dst, int n) {
    int i = (blockIdx.x * blockDim.x + threadIdx.x) * 4;
    if (i < n) {
        f4 v = *(const f4*)(src + i);
        ush4 o;
        o.x = f2bf(v.x); o.y = f2bf(v.y); o.z = f2bf(v.z); o.w = f2bf(v.w);
        *(ush4*)(dst + i) = o;
    }
}

// ---------------- generic bf16 GEMM: C = A(MxK) * B(KxN) ----------------
// block tile 256(M) x 64(N), 256 threads = 8 waves; wave owns 32x64 output.
// Double-buffered LDS; A tile staged by TDM (async, TENSORcnt), B tile staged
// by pipelined global loads + transposed ds stores. One barrier per K-step.
// mode 0: fp32 out; mode 1: scatter bf16 q/k/v (B,H,T,DH).
__global__ __launch_bounds__(256)
void mha_gemm_bf16(const unsigned short* __restrict__ A,
                   const unsigned short* __restrict__ Bm,
                   int M, int N, int K,
                   float* __restrict__ outF,
                   unsigned short* __restrict__ qd,
                   unsigned short* __restrict__ kd,
                   unsigned short* __restrict__ vd,
                   int mode) {
    __shared__ unsigned short ldsA[2][256 * 32];   // [m][k]  2 x 16 KB
    __shared__ unsigned short ldsBt[2][64 * 32];   // [n][k]  2 x  4 KB

    const int tid  = threadIdx.x;
    const int wid  = tid >> 5;
    const int lane = tid & 31;
    const int lh   = lane & 15;
    const int hi   = lane >> 4;

    const int m0 = blockIdx.x * 256;
    const int n0 = blockIdx.y * 64;

    const int kr = tid >> 3;            // B staging: k row 0..31
    const int nc = (tid & 7) * 8;       // B staging: n chunk

    v8f acc[2][4] = {};

    // ---- prologue: stage tile 0 into buffer 0 ----
    if (wid == 0)
        tdm_load_2d(lds_off32(&ldsA[0][0]),
                    (unsigned long long)(A + (size_t)m0 * K),
                    (unsigned)K, (unsigned)M, 32u, 256u, (unsigned long long)K);
    {
        const unsigned short* g = Bm + (size_t)kr * N + n0 + nc;
        ush8 bv = *(const ush8*)g;
        #pragma unroll
        for (int e = 0; e < 8; ++e)
            ldsBt[0][(nc + e) * 32 + kr] = bv[e];
    }

    int p = 0;
    for (int k0 = 0; k0 < K; k0 += 32, p ^= 1) {
        if (wid == 0)
            __builtin_amdgcn_s_wait_tensorcnt(0);   // TDM for buffer p done
        __syncthreads();                            // buffer p visible to all

        const bool more = (k0 + 32) < K;
        ush8 bnext;
        if (more) {
            if (wid == 0)                           // async-stage A tile k0+32
                tdm_load_2d(lds_off32(&ldsA[p ^ 1][0]),
                            (unsigned long long)(A + (size_t)m0 * K + k0 + 32),
                            (unsigned)K, (unsigned)M, 32u, 256u,
                            (unsigned long long)K);
            bnext = *(const ush8*)(Bm + (size_t)(k0 + 32 + kr) * N + n0 + nc);
        }

        // ---- compute from buffer p: 8 WMMAs ----
        FragBF16 af[2];
        #pragma unroll
        for (int r = 0; r < 2; ++r) {
            const unsigned short* ab = &ldsA[p][(wid * 32 + r * 16 + lh) * 32];
            af[r].u[0] = *(const ush8*)(ab + hi * 8);       // K 0-7 / 8-15
            af[r].u[1] = *(const ush8*)(ab + 16 + hi * 8);  // K 16-23 / 24-31
        }
        #pragma unroll
        for (int j = 0; j < 4; ++j) {
            FragBF16 bf_;
            const unsigned short* bb = &ldsBt[p][(j * 16 + lh) * 32 + hi * 8];
            bf_.u[0] = *(const ush8*)bb;
            bf_.u[1] = *(const ush8*)(bb + 16);
            #pragma unroll
            for (int r = 0; r < 2; ++r)
                acc[r][j] = __builtin_amdgcn_wmma_f32_16x16x32_bf16(
                    false, af[r].b, false, bf_.b, (short)0, acc[r][j], false, false);
        }

        // ---- drain pipelined B tile into back buffer ----
        if (more) {
            #pragma unroll
            for (int e = 0; e < 8; ++e)
                ldsBt[p ^ 1][(nc + e) * 32 + kr] = bnext[e];
        }
    }

    // ---- epilogue ----
    #pragma unroll
    for (int r = 0; r < 2; ++r) {
        #pragma unroll
        for (int j = 0; j < 4; ++j) {
            #pragma unroll
            for (int i = 0; i < 8; ++i) {
                float val = acc[r][j][i];
                int row = m0 + wid * 32 + r * 16 + i + hi * 8; // C: lanes>=16 -> +8
                int col = n0 + j * 16 + lh;
                if (mode == 0) {
                    outF[(size_t)row * N + col] = val;
                } else {
                    int sel = col >> 10;                   // 0=Q 1=K 2=V
                    int rr  = col & 1023;
                    int h   = rr >> 6;
                    int dh  = rr & 63;
                    int b   = row >> 11;
                    int t   = row & 2047;
                    unsigned short* dst = (sel == 0) ? qd : (sel == 1) ? kd : vd;
                    dst[(((size_t)b * NHEAD + h) * SEQ + t) * DHEAD + dh] = f2bf(val);
                }
            }
        }
    }
}

// ---------------- flash attention over (B*H) heads ----------------
// block: 128 threads = 4 waves; 64 q rows per block, 64 key cols per step.
// K tile staged by TDM (contiguous 64x64 rows of this head), V tile staged by
// pipelined global loads + transposed ds stores; double-buffered LDS.
__global__ __launch_bounds__(128)
void mha_flash_attn(const unsigned short* __restrict__ qd,
                    const unsigned short* __restrict__ kd,
                    const unsigned short* __restrict__ vd,
                    unsigned short* __restrict__ ctx) {
    __shared__ unsigned short ldsK[2][64 * 64];     // [t][dh]  2 x 8 KB
    __shared__ unsigned short ldsVt[2][64 * 64];    // [dh][t]  2 x 8 KB
    __shared__ unsigned short ldsP[4][16 * 64];     // per-wave P tile  8 KB

    const int tid  = threadIdx.x;
    const int wid  = tid >> 5;
    const int lane = tid & 31;
    const int lh   = lane & 15;
    const int hi   = lane >> 4;

    const int bh = blockIdx.x;            // b*H + h
    const int b  = bh >> 4;
    const int h  = bh & 15;
    const int q0 = blockIdx.y * 64;

    const int vrow  = tid >> 1;           // V staging: t row 0..63
    const int vhalf = (tid & 1) * 32;     // V staging: dh half

    // Q fragments for this wave's 16 rows (dh split into two K=32 steps)
    const unsigned short* qbase =
        qd + ((size_t)bh * SEQ + q0 + wid * 16 + lh) * DHEAD;
    FragBF16 qf[2];
    #pragma unroll
    for (int s = 0; s < 2; ++s) {
        qf[s].u[0] = *(const ush8*)(qbase + s * 32 + hi * 8);
        qf[s].u[1] = *(const ush8*)(qbase + s * 32 + 16 + hi * 8);
    }

    float mrow[8], lrow[8];
    #pragma unroll
    for (int i = 0; i < 8; ++i) { mrow[i] = -1e30f; lrow[i] = 0.f; }
    v8f acc[4] = {};

    const float scale = 0.125f;   // 1/sqrt(64)

    // ---- prologue: stage tile kt=0 into buffer 0 ----
    if (wid == 0)
        tdm_load_2d(lds_off32(&ldsK[0][0]),
                    (unsigned long long)(kd + (size_t)bh * SEQ * DHEAD),
                    64u, (unsigned)SEQ, 64u, 64u, 64ull);
    {
        const unsigned short* g =
            vd + ((size_t)bh * SEQ + vrow) * DHEAD + vhalf;
        #pragma unroll
        for (int c = 0; c < 4; ++c) {
            ush8 v0 = *(const ush8*)(g + c * 8);
            #pragma unroll
            for (int e = 0; e < 8; ++e)
                ldsVt[0][(vhalf + c * 8 + e) * 64 + vrow] = v0[e];
        }
    }

    int p = 0;
    for (int kt = 0; kt < SEQ; kt += 64, p ^= 1) {
        if (wid == 0)
            __builtin_amdgcn_s_wait_tensorcnt(0);   // K tile for buffer p done
        __syncthreads();                            // buffer p visible

        const bool more = (kt + 64) < SEQ;
        ush8 vnext[4];
        if (more) {
            if (wid == 0)
                tdm_load_2d(lds_off32(&ldsK[p ^ 1][0]),
                            (unsigned long long)(kd + ((size_t)bh * SEQ + kt + 64) * DHEAD),
                            64u, (unsigned)SEQ, 64u, 64u, 64ull);
            const unsigned short* g =
                vd + ((size_t)bh * SEQ + kt + 64 + vrow) * DHEAD + vhalf;
            #pragma unroll
            for (int c = 0; c < 4; ++c)
                vnext[c] = *(const ush8*)(g + c * 8);
        }

        // ---- S = Q * K^T for 64 key columns (four 16x16 tiles) ----
        v8f S[4] = {};
        #pragma unroll
        for (int ct = 0; ct < 4; ++ct) {
            #pragma unroll
            for (int s = 0; s < 2; ++s) {
                FragBF16 kf;
                const unsigned short* kb =
                    &ldsK[p][(ct * 16 + lh) * 64 + s * 32 + hi * 8];
                kf.u[0] = *(const ush8*)kb;
                kf.u[1] = *(const ush8*)(kb + 16);
                S[ct] = __builtin_amdgcn_wmma_f32_16x16x32_bf16(
                    false, qf[s].b, false, kf.b, (short)0, S[ct], false, false);
            }
        }

        // ---- online softmax over the 64 columns ----
        float corr[8];
        #pragma unroll
        for (int i = 0; i < 8; ++i) {
            float sv[4];
            #pragma unroll
            for (int ct = 0; ct < 4; ++ct) sv[ct] = S[ct][i] * scale;
            float mx = fmaxf(fmaxf(sv[0], sv[1]), fmaxf(sv[2], sv[3]));
            #pragma unroll
            for (int msk = 1; msk < 16; msk <<= 1)
                mx = fmaxf(mx, __shfl_xor(mx, msk, 32));
            float mnew = fmaxf(mrow[i], mx);
            float pv[4], rs = 0.f;
            #pragma unroll
            for (int ct = 0; ct < 4; ++ct) { pv[ct] = __expf(sv[ct] - mnew); rs += pv[ct]; }
            #pragma unroll
            for (int msk = 1; msk < 16; msk <<= 1)
                rs += __shfl_xor(rs, msk, 32);
            float c = __expf(mrow[i] - mnew);
            lrow[i] = lrow[i] * c + rs;
            mrow[i] = mnew;
            corr[i] = c;
            int prow = i + hi * 8;
            #pragma unroll
            for (int ct = 0; ct < 4; ++ct)
                ldsP[wid][prow * 64 + ct * 16 + lh] = f2bf(pv[ct]);
        }
        #pragma unroll
        for (int j = 0; j < 4; ++j)
            #pragma unroll
            for (int i = 0; i < 8; ++i)
                acc[j][i] *= corr[i];

        // per-wave LDS turnaround: DS ops in a wave are in-order; wait for data
        asm volatile("s_wait_dscnt 0" ::: "memory");

        // ---- ctx += P(16x64) * V(64x64): two K=32 steps x four N-tiles ----
        FragBF16 pf[2];
        #pragma unroll
        for (int s = 0; s < 2; ++s) {
            const unsigned short* pb = &ldsP[wid][lh * 64 + s * 32];
            pf[s].u[0] = *(const ush8*)(pb + hi * 8);
            pf[s].u[1] = *(const ush8*)(pb + 16 + hi * 8);
        }
        #pragma unroll
        for (int j = 0; j < 4; ++j) {
            #pragma unroll
            for (int s = 0; s < 2; ++s) {
                FragBF16 vf;
                const unsigned short* vb =
                    &ldsVt[p][(j * 16 + lh) * 64 + s * 32 + hi * 8];
                vf.u[0] = *(const ush8*)vb;
                vf.u[1] = *(const ush8*)(vb + 16);
                acc[j] = __builtin_amdgcn_wmma_f32_16x16x32_bf16(
                    false, pf[s].b, false, vf.b, (short)0, acc[j], false, false);
            }
        }

        // ---- drain pipelined V tile into back buffer ----
        if (more) {
            #pragma unroll
            for (int c = 0; c < 4; ++c)
                #pragma unroll
                for (int e = 0; e < 8; ++e)
                    ldsVt[p ^ 1][(vhalf + c * 8 + e) * 64 + vrow] = vnext[c][e];
        }
    }

    // ---- normalize and store ctx (B,T,D) bf16 ----
    #pragma unroll
    for (int i = 0; i < 8; ++i) {
        float inv = 1.f / lrow[i];
        int t = q0 + wid * 16 + i + hi * 8;
        #pragma unroll
        for (int j = 0; j < 4; ++j) {
            int dh = j * 16 + lh;
            ctx[((size_t)b * SEQ + t) * DMODEL + h * DHEAD + dh] =
                f2bf(acc[j][i] * inv);
        }
    }
}

// ---------------- host-side launch ----------------
extern "C" void kernel_launch(void* const* d_in, const int* in_sizes, int n_in,
                              void* d_out, int out_size, void* d_ws, size_t ws_size,
                              hipStream_t stream) {
    const float* x    = (const float*)d_in[0];   // (4,2048,1024)
    const float* Wqkv = (const float*)d_in[1];   // (1024,3072)
    const float* Wout = (const float*)d_in[2];   // (1024,1024)
    float* out = (float*)d_out;                  // (4,2048,1024)

    char* ws = (char*)d_ws;
    size_t off = 0;
    unsigned short* xb    = (unsigned short*)(ws + off); off += (size_t)MROWS * DMODEL * 2;       // 16 MiB
    unsigned short* wqkvb = (unsigned short*)(ws + off); off += (size_t)DMODEL * 3 * DMODEL * 2;  //  6 MiB
    unsigned short* woutb = (unsigned short*)(ws + off); off += (size_t)DMODEL * DMODEL * 2;      //  2 MiB
    unsigned short* qb    = (unsigned short*)(ws + off); off += (size_t)MROWS * DMODEL * 2;       // 16 MiB
    unsigned short* kb    = (unsigned short*)(ws + off); off += (size_t)MROWS * DMODEL * 2;       // 16 MiB
    unsigned short* vb    = (unsigned short*)(ws + off); off += (size_t)MROWS * DMODEL * 2;       // 16 MiB
    unsigned short* ctxb  = (unsigned short*)(ws + off); off += (size_t)MROWS * DMODEL * 2;       // 16 MiB

    // 1) convert inputs to bf16
    {
        int n = MROWS * DMODEL;                       // 8388608
        mha_cvt_bf16<<<n / 4 / 256, 256, 0, stream>>>(x, xb, n);
        n = DMODEL * 3 * DMODEL;                      // 3145728
        mha_cvt_bf16<<<n / 4 / 256, 256, 0, stream>>>(Wqkv, wqkvb, n);
        n = DMODEL * DMODEL;                          // 1048576
        mha_cvt_bf16<<<n / 4 / 256, 256, 0, stream>>>(Wout, woutb, n);
    }

    // 2) QKV projection + scatter to (B,H,T,DH)
    mha_gemm_bf16<<<dim3(MROWS / 256, (3 * DMODEL) / 64), 256, 0, stream>>>(
        xb, wqkvb, MROWS, 3 * DMODEL, DMODEL, nullptr, qb, kb, vb, 1);

    // 3) flash attention -> ctx (B,T,D) bf16
    mha_flash_attn<<<dim3(BATCH * NHEAD, SEQ / 64), 128, 0, stream>>>(qb, kb, vb, ctxb);

    // 4) output projection -> fp32 out
    mha_gemm_bf16<<<dim3(MROWS / 256, DMODEL / 64), 256, 0, stream>>>(
        ctxb, woutb, MROWS, DMODEL, DMODEL, out, nullptr, nullptr, nullptr, 0);
}
